// sLSTMBlock_20461224198601
// MI455X (gfx1250) — matrix operand order repaired and checked by reading, not compiled, not run on hardware
//
#include <hip/hip_runtime.h>
#include <hip/hip_bf16.h>
#include <cstdint>

typedef __attribute__((ext_vector_type(16))) _Float16 v16h;
typedef __attribute__((ext_vector_type(8)))  float    v8f;
typedef unsigned int u32;

union FragU { u32 u[8]; v16h v; };

#define NB    8        // batch
#define SS    4096     // sequence
#define DD    1024     // model dim
#define NN    64       // head dim
#define GCOLS 256      // 4 gates * 64
#define KT_G  32       // 1024/32 k-steps, gate GEMM
#define KT_O  2        // 64/32  k-steps, out GEMM
#define CHUNK 64
#define NCHUNK 64      // 4096/64
#define NCHAN 512      // B*N channels

// ---- workspace layout (bytes) ----
#define WS_WB    ((size_t)0)                                    // 32*16*32*16 f16 = 512 KB
#define WS_WPB   ((size_t)(512*1024))                           // 2*64*32*16 f16 = 128 KB
#define WS_GATES ((size_t)(640*1024))                           // S*B*256 f32 = 32 MB
#define WS_LF    (WS_GATES + (size_t)SS*NB*GCOLS*4)
#define WS_ML    (WS_LF  + (size_t)NCHUNK*NCHAN*4)
#define WS_CL    (WS_ML  + (size_t)NCHUNK*NCHAN*4)
#define WS_CIN   (WS_CL  + (size_t)NCHUNK*NCHAN*4)
#define WS_MIN   (WS_CIN + (size_t)NCHUNK*NCHAN*4)
#define WS_H     (WS_MIN + (size_t)NCHUNK*NCHAN*4)              // 32768*64 f16 = 4 MB

// ---------------------------------------------------------------------------
// Pack 4 gate weight matrices (each N x D f32, row-major) into one fused f16
// B-operand, fragment-major: idx = ((kt*16 + ct)*32 + lane)*16 + j
// where col = ct*16 + (lane&15) in [0,256), k = kt*32 + (j>>3)*16 + (lane>>4)*8 + (j&7).
// ---------------------------------------------------------------------------
__global__ void pack_gate_w_kernel(const float* __restrict__ Wi, const float* __restrict__ Wf,
                                   const float* __restrict__ Wz, const float* __restrict__ Wo,
                                   _Float16* __restrict__ WB) {
    u32 t = blockIdx.x * blockDim.x + threadIdx.x;          // 2^18 total
    u32 j    =  t        & 15;
    u32 lane = (t >> 4)  & 31;
    u32 ct   = (t >> 9)  & 15;
    u32 kt   =  t >> 13;                                    // 0..31
    u32 col  = ct * 16 + (lane & 15);
    u32 k    = kt * 32 + ((j >> 3) << 4) + ((lane >> 4) << 3) + (j & 7);
    u32 gate = col >> 6, n = col & 63;
    const float* W = (gate == 0) ? Wi : (gate == 1) ? Wf : (gate == 2) ? Wz : Wo;
    WB[t] = (_Float16)W[(size_t)n * DD + k];
}

// Wp is (D, N) f32. Fused out B-operand: idx = ((kt*64 + ct)*32 + lane)*16 + j
// col(d) = ct*16 + (lane&15), k(n) = kt*32 + (j>>3)*16 + (lane>>4)*8 + (j&7).
__global__ void pack_out_w_kernel(const float* __restrict__ Wp, _Float16* __restrict__ WPB) {
    u32 t = blockIdx.x * blockDim.x + threadIdx.x;          // 2^16 total
    u32 j    =  t        & 15;
    u32 lane = (t >> 4)  & 31;
    u32 ct   = (t >> 9)  & 63;
    u32 kt   =  t >> 15;                                    // 0..1
    u32 d    = ct * 16 + (lane & 15);
    u32 k    = kt * 32 + ((j >> 3) << 4) + ((lane >> 4) << 3) + (j & 7);
    WPB[t] = (_Float16)Wp[(size_t)d * NN + k];
}

// ---------------------------------------------------------------------------
// Fused LayerNorm + 4-gate projection.  One 16-token tile per workgroup,
// 256 threads (8 waves).  WMMA f16 -> f32, fused activation epilogue.
// gates layout: gates[(s*8 + b)*256 + gate*64 + n]  (time-major for the scan)
// ---------------------------------------------------------------------------
__global__ void __launch_bounds__(256)
ln_gate_kernel(const float* __restrict__ x, const float* __restrict__ ln_w,
               const float* __restrict__ ln_b, const _Float16* __restrict__ WB,
               const float* __restrict__ bi, const float* __restrict__ bfv,
               const float* __restrict__ bz, const float* __restrict__ bo,
               float* __restrict__ gates) {
    __shared__ _Float16 sA[16][DD + 8];                     // +8 halves: kill LDS bank conflicts

    const int wave = threadIdx.x >> 5;
    const int lane = threadIdx.x & 31;

    // ---- LayerNorm: each wave normalizes 2 rows of the 16-row tile ----
    #pragma unroll
    for (int rr = 0; rr < 2; ++rr) {
        const int m = wave * 2 + rr;
        const size_t r = (size_t)blockIdx.x * 16 + m;       // token id, b = r>>12, s = r&4095
        const float* xr = x + r * DD;
        float xv[32];
        float s0 = 0.f, s1 = 0.f;
        #pragma unroll
        for (int i = 0; i < 32; ++i) {
            float v = xr[i * 32 + lane];
            xv[i] = v; s0 += v; s1 = fmaf(v, v, s1);
        }
        #pragma unroll
        for (int off = 16; off > 0; off >>= 1) {
            s0 += __shfl_xor(s0, off);
            s1 += __shfl_xor(s1, off);
        }
        float mean = s0 * (1.0f / DD);
        float var  = s1 * (1.0f / DD) - mean * mean;
        float rstd = rsqrtf(var + 1e-5f);
        #pragma unroll
        for (int i = 0; i < 32; ++i) {
            int k = i * 32 + lane;
            float v = (xv[i] - mean) * rstd * ln_w[k] + ln_b[k];
            sA[m][k] = (_Float16)v;
        }
    }
    __syncthreads();

    // ---- GEMM: wave w owns column tiles {2w, 2w+1} of 16 cols each ----
    const int hi   = lane >> 4;      // 0: lanes 0-15, 1: lanes 16-31
    const int mrow = lane & 15;
    v8f acc[2];
    #pragma unroll
    for (int q = 0; q < 2; ++q)
        #pragma unroll
        for (int v = 0; v < 8; ++v) acc[q][v] = 0.0f;

    for (int kt = 0; kt < KT_G; ++kt) {
        FragU a;
        #pragma unroll
        for (int v = 0; v < 8; ++v) {                       // ISA 16-bit A layout
            int k = kt * 32 + ((v >> 2) << 4) + (hi << 3) + ((v & 3) << 1);
            a.u[v] = *(const u32*)&sA[mrow][k];
        }
        if (kt + 1 < KT_G)                                  // CDNA5 global_prefetch_b8 (near scope)
            __builtin_prefetch((const void*)(WB + ((size_t)((kt + 1) * 16 + wave * 2) * 32 + lane) * 16), 0, 3);
        #pragma unroll
        for (int q = 0; q < 2; ++q) {
            int ct = wave * 2 + q;
            const uint4* bp4 = (const uint4*)(WB + ((size_t)(kt * 16 + ct) * 32 + lane) * 16);
            uint4 b0 = bp4[0], b1 = bp4[1];
            FragU bf;
            bf.u[0] = b0.x; bf.u[1] = b0.y; bf.u[2] = b0.z; bf.u[3] = b0.w;
            bf.u[4] = b1.x; bf.u[5] = b1.y; bf.u[6] = b1.z; bf.u[7] = b1.w;
            acc[q] = __builtin_amdgcn_wmma_f32_16x16x32_f16(
                false, a.v, false, bf.v, (short)0, acc[q], false, false);
        }
    }

    // ---- epilogue: bias + per-gate activation (gate uniform per col-tile) ----
    #pragma unroll
    for (int q = 0; q < 2; ++q) {
        int ct   = wave * 2 + q;
        int col  = ct * 16 + mrow;
        int gate = ct >> 2;                                 // 4 col-tiles per gate
        int n    = col & 63;
        float bias = (gate == 0) ? bi[n] : (gate == 1) ? bfv[n] : (gate == 2) ? bz[n] : bo[n];
        #pragma unroll
        for (int v = 0; v < 8; ++v) {
            int m = hi * 8 + v;
            size_t r = (size_t)blockIdx.x * 16 + m;
            int b_ = (int)(r >> 12), s = (int)(r & 4095);
            float val = acc[q][v] + bias;
            if (gate < 2)       val = fminf(fmaxf(val, -20.f), 20.f);
            else if (gate == 2) val = tanhf(val);
            else                val = 1.0f / (1.0f + __expf(-val));
            gates[((size_t)(s * NB + b_)) * GCOLS + col] = val;
        }
    }
}

// ---------------------------------------------------------------------------
// Chunked stabilized scan.  State (c, m); true value C = c * exp(m).
// Pass 1: per (channel, chunk) compute Lf = sum(lf), (Mloc, cloc) from
//         (cloc, Mloc) = (0, -inf).
// ---------------------------------------------------------------------------
__global__ void scan_pass1_kernel(const float* __restrict__ gates,
                                  float* __restrict__ Lf, float* __restrict__ Ml,
                                  float* __restrict__ Cl) {
    u32 t = blockIdx.x * blockDim.x + threadIdx.x;          // NCHUNK*NCHAN = 32768
    u32 chunk = t >> 9, l = t & 511;
    u32 b = l >> 6, n = l & 63;
    float lfSum = 0.f, m = -1e30f, c = 0.f;
    for (int i = 0; i < CHUNK; ++i) {
        int s = chunk * CHUNK + i;
        size_t base = ((size_t)(s * NB + b)) * GCOLS;
        float li = gates[base + n];
        float lf = gates[base + 64 + n];
        float z  = gates[base + 128 + n];
        lfSum += lf;
        float mn = fmaxf(m + lf, li);
        c = __expf(m + lf - mn) * c + __expf(li - mn) * z;
        m = mn;
    }
    Lf[t] = lfSum; Ml[t] = m; Cl[t] = c;
}

// Pass 2: per channel, serially combine 64 chunk summaries; record the
// incoming (c, m) for every chunk.  m_out = max(m_in+Lf, Mloc),
// c_out = exp(m_in+Lf-m_out)*c_in + exp(Mloc-m_out)*cloc.
__global__ void scan_pass2_kernel(const float* __restrict__ Lf, const float* __restrict__ Ml,
                                  const float* __restrict__ Cl,
                                  float* __restrict__ c_in, float* __restrict__ m_in) {
    u32 l = blockIdx.x * blockDim.x + threadIdx.x;          // 512 channels
    if (l >= NCHAN) return;
    float c = 0.f, m = 0.f;                                 // reference init c0=0, m0=0
    for (int j = 0; j < NCHUNK; ++j) {
        u32 idx = j * NCHAN + l;
        c_in[idx] = c; m_in[idx] = m;
        float mo = fmaxf(m + Lf[idx], Ml[idx]);
        c = __expf(m + Lf[idx] - mo) * c + __expf(Ml[idx] - mo) * Cl[idx];
        m = mo;
    }
}

// Pass 3: replay the exact reference recursion per chunk from the correct
// incoming state; emit h = o * tanh(c) as f16 in (token, n) row-major order.
__global__ void scan_pass3_kernel(const float* __restrict__ gates,
                                  const float* __restrict__ c_in, const float* __restrict__ m_in,
                                  _Float16* __restrict__ h) {
    u32 t = blockIdx.x * blockDim.x + threadIdx.x;
    u32 chunk = t >> 9, l = t & 511;
    u32 b = l >> 6, n = l & 63;
    float c = c_in[chunk * NCHAN + l];
    float m = m_in[chunk * NCHAN + l];
    for (int i = 0; i < CHUNK; ++i) {
        int s = chunk * CHUNK + i;
        size_t base = ((size_t)(s * NB + b)) * GCOLS;
        float li = gates[base + n];
        float lf = gates[base + 64 + n];
        float z  = gates[base + 128 + n];
        float o  = gates[base + 192 + n];
        float mn = fmaxf(m + lf, li);
        c = __expf(m + lf - mn) * c + __expf(li - mn) * z;
        m = mn;
        h[((size_t)b * SS + s) * NN + n] = (_Float16)(o * tanhf(c));
    }
}

// ---------------------------------------------------------------------------
// Output projection: h (32768 x 64 f16) x WpB -> out (32768 x 1024 f32) + bp.
// One 16-token tile per workgroup; 8 waves x 8 col-tiles; K = 64 (2 WMMAs).
// h tile staged via CDNA5 async memory->LDS DMA path (ASYNCcnt).
// ---------------------------------------------------------------------------
__global__ void __launch_bounds__(256)
out_gemm_kernel(const _Float16* __restrict__ h, const _Float16* __restrict__ WPB,
                const float* __restrict__ bp, float* __restrict__ out) {
    __shared__ _Float16 sH[16][NN + 8];                     // row stride 144 B (16B-aligned)

    const int wave = threadIdx.x >> 5;
    const int lane = threadIdx.x & 31;
    const int hi   = lane >> 4;
    const int mrow = lane & 15;

    // ---- async-stage the 16x64 h tile (2 KB) straight into LDS ----
    // 128 lanes x one GLOBAL_LOAD_ASYNC_TO_LDS_B128 transfer (16 B each).
    if (threadIdx.x < 128) {
        int row = threadIdx.x >> 3;                         // 0..15
        int seg = threadIdx.x & 7;                          // 8 x 16 B per row
        const _Float16* gsrc = h + (size_t)blockIdx.x * 16 * NN + (size_t)row * NN + seg * 8;
        u32 ldsdst = (u32)(uintptr_t)&sH[0][0] + (u32)(row * (NN + 8) + seg * 8) * 2;
        asm volatile("global_load_async_to_lds_b128 %0, %1, off"
                     :: "v"(ldsdst), "v"(gsrc) : "memory");
    }
    asm volatile("s_wait_asynccnt 0x0" ::: "memory");
    __syncthreads();

    v8f acc[8];
    #pragma unroll
    for (int q = 0; q < 8; ++q)
        #pragma unroll
        for (int v = 0; v < 8; ++v) acc[q][v] = 0.0f;

    #pragma unroll
    for (int kt = 0; kt < KT_O; ++kt) {
        FragU a;
        #pragma unroll
        for (int v = 0; v < 8; ++v) {
            int k = kt * 32 + ((v >> 2) << 4) + (hi << 3) + ((v & 3) << 1);
            a.u[v] = *(const u32*)&sH[mrow][k];
        }
        #pragma unroll
        for (int q = 0; q < 8; ++q) {
            int ct = wave * 8 + q;
            const uint4* bp4 = (const uint4*)(WPB + ((size_t)(kt * 64 + ct) * 32 + lane) * 16);
            uint4 b0 = bp4[0], b1 = bp4[1];
            FragU bf;
            bf.u[0] = b0.x; bf.u[1] = b0.y; bf.u[2] = b0.z; bf.u[3] = b0.w;
            bf.u[4] = b1.x; bf.u[5] = b1.y; bf.u[6] = b1.z; bf.u[7] = b1.w;
            acc[q] = __builtin_amdgcn_wmma_f32_16x16x32_f16(
                false, a.v, false, bf.v, (short)0, acc[q], false, false);
        }
    }

    #pragma unroll
    for (int q = 0; q < 8; ++q) {
        int ct  = wave * 8 + q;
        int col = ct * 16 + mrow;
        float bias = bp[col];
        #pragma unroll
        for (int v = 0; v < 8; ++v) {
            int m = hi * 8 + v;
            size_t r = (size_t)blockIdx.x * 16 + m;
            out[r * DD + col] = acc[q][v] + bias;
        }
    }
}

// ---------------------------------------------------------------------------
extern "C" void kernel_launch(void* const* d_in, const int* in_sizes, int n_in,
                              void* d_out, int out_size, void* d_ws, size_t ws_size,
                              hipStream_t stream) {
    const float* x    = (const float*)d_in[0];
    const float* ln_w = (const float*)d_in[1];
    const float* ln_b = (const float*)d_in[2];
    const float* Wi   = (const float*)d_in[3];
    const float* bi   = (const float*)d_in[4];
    const float* Wf   = (const float*)d_in[5];
    const float* bfv  = (const float*)d_in[6];
    const float* Wz   = (const float*)d_in[7];
    const float* bz   = (const float*)d_in[8];
    const float* Wo   = (const float*)d_in[9];
    const float* bo   = (const float*)d_in[10];
    const float* Wp   = (const float*)d_in[11];
    const float* bp   = (const float*)d_in[12];
    float* out = (float*)d_out;

    char* ws = (char*)d_ws;
    _Float16* WB    = (_Float16*)(ws + WS_WB);
    _Float16* WPB   = (_Float16*)(ws + WS_WPB);
    float*    gates = (float*)   (ws + WS_GATES);
    float*    Lf    = (float*)   (ws + WS_LF);
    float*    Ml    = (float*)   (ws + WS_ML);
    float*    Cl    = (float*)   (ws + WS_CL);
    float*    c_in  = (float*)   (ws + WS_CIN);
    float*    m_in  = (float*)   (ws + WS_MIN);
    _Float16* hbuf  = (_Float16*)(ws + WS_H);

    const int rowTiles = (NB * SS) / 16;                    // 2048

    pack_gate_w_kernel<<<dim3(1024), dim3(256), 0, stream>>>(Wi, Wf, Wz, Wo, WB);
    pack_out_w_kernel <<<dim3(256),  dim3(256), 0, stream>>>(Wp, WPB);
    ln_gate_kernel    <<<dim3(rowTiles), dim3(256), 0, stream>>>(x, ln_w, ln_b, WB,
                                                                 bi, bfv, bz, bo, gates);
    scan_pass1_kernel <<<dim3((NCHUNK * NCHAN) / 256), dim3(256), 0, stream>>>(gates, Lf, Ml, Cl);
    scan_pass2_kernel <<<dim3(2), dim3(256), 0, stream>>>(Lf, Ml, Cl, c_in, m_in);
    scan_pass3_kernel <<<dim3((NCHUNK * NCHAN) / 256), dim3(256), 0, stream>>>(gates, c_in, m_in, hbuf);
    out_gemm_kernel   <<<dim3(rowTiles), dim3(256), 0, stream>>>(hbuf, WPB, bp, out);
}
